// SwinModel_28733331210804
// MI455X (gfx1250) — matrix operand order repaired and checked by reading, not compiled, not run on hardware
//
#include <hip/hip_runtime.h>
#include <hip/hip_bf16.h>
#include <cmath>

// ---------------------------------------------------------------------------
// CDNA5 (gfx1250) Swin-UNet forward.  All linear layers go through
// v_wmma_f32_16x16x32_f16 (f16 inputs, f32 accumulate).  wave32 everywhere.
// ---------------------------------------------------------------------------

typedef __attribute__((ext_vector_type(16))) _Float16 v16h;
typedef __attribute__((ext_vector_type(8)))  float    v8f;
typedef __attribute__((ext_vector_type(4)))  float    v4f;

static __device__ __forceinline__ float gelu_exact(float x) {
  return 0.5f * x * (1.0f + erff(x * 0.70710678118654752440f));
}
static __device__ __forceinline__ float wave_sum(float v) {
#pragma unroll
  for (int o = 16; o > 0; o >>= 1) v += __shfl_xor(v, o, 32);
  return v;
}
static __device__ __forceinline__ v16h cvt16(v4f y0, v4f y1, v4f y2, v4f y3) {
  v16h b;
#pragma unroll
  for (int e = 0; e < 4; ++e) {
    b[e]      = (_Float16)y0[e];
    b[e + 4]  = (_Float16)y1[e];
    b[e + 8]  = (_Float16)y2[e];
    b[e + 12] = (_Float16)y3[e];
  }
  return b;
}

// ---------------------------------------------------------------------------
// Generic fused GEMM:  Out[M,N] = act( A[M,K] * W[N,K]^T + bias[N] + res )
// One wave computes a 32x64 strip: 2 A fragments x 4 B fragments -> 8 WMMAs
// per 32-wide K step.  Wave id is pinned to SGPRs via readfirstlane so all
// tile guards are scalar branches (no EXEC churn).  Loads are never
// predicated: row/col pointers are clamped (stores keep the bounds check),
// the K tail (patch embed, K=48) uses address-clamped loads + select and only
// exists in the !FULLN instantiation.
// Fragment layouts per CDNA5 ISA 7.12.2:
//   A 16x32 f16: lane L -> row = L%16;  elem e -> K = (e<8?e:e+8) + 8*(L/16)
//   B 32x16 f16: lane L -> col = L%16;  elem e -> K = (e<8?e:e+8) + 8*(L/16)
//   D 16x16 f32: lane L -> col = L%16;  vgpr r -> row = r + 8*(L/16)
// ---------------------------------------------------------------------------
template <bool FULLN>
__global__ __launch_bounds__(256) void k_gemm_wmma(
    const float* __restrict__ A, const float* __restrict__ Wt,
    const float* __restrict__ bias, const float* __restrict__ res,
    float* __restrict__ Out, int M, int N, int K,
    int ldA, int ldO, int ldR, int act)
{
  int lane = threadIdx.x & 31;
  int wid  = __builtin_amdgcn_readfirstlane(threadIdx.x >> 5);   // SGPR
  int tilesN4 = (N + 63) >> 6;
  int tilesM2 = (M + 31) >> 5;
  int tile = blockIdx.x * 8 + wid;                               // scalar
  if (tile >= tilesM2 * tilesN4) return;                         // s_cbranch
  int tm = tile / tilesN4, tn = tile % tilesN4;
  int half = lane >> 4;
  int l16  = lane & 15;
  int cbase = tn * 64;                                           // scalar

  int row0 = tm * 32 + l16;
  int row1 = row0 + 16;
  int r0c = (row0 < M) ? row0 : (M - 1);      // clamp; stores stay guarded
  int r1c = (row1 < M) ? row1 : (M - 1);
  const float* Arow0 = A + (size_t)r0c * ldA;
  const float* Arow1 = A + (size_t)r1c * ldA;

  const float* Wr[4];
#pragma unroll
  for (int j = 0; j < 4; ++j) {
    int wn  = cbase + j * 16 + l16;
    int wnc = (wn < N) ? wn : (N - 1);        // clamp
    Wr[j] = Wt + (size_t)wnc * K;
  }

  v8f accA[4] = {{}, {}, {}, {}};
  v8f accB[4] = {{}, {}, {}, {}};
  int kfull = K & ~31;
  int k0 = 0;
  for (; k0 < kfull; k0 += 32) {
    __builtin_prefetch(Arow0 + k0 + 64, 0, 1);   // global_prefetch_b8 (A in ws)
    // ---- issue ALL loads first (one clause, up to 24 b128 in flight) ----
    const v4f* pa0 = (const v4f*)(Arow0 + k0 + 8 * half);
    const v4f* pa1 = (const v4f*)(Arow0 + k0 + 16 + 8 * half);
    v4f ax0 = pa0[0], ax1 = pa0[1], ax2 = pa1[0], ax3 = pa1[1];
    const v4f* pb0 = (const v4f*)(Arow1 + k0 + 8 * half);
    const v4f* pb1 = (const v4f*)(Arow1 + k0 + 16 + 8 * half);
    v4f bx0 = pb0[0], bx1 = pb0[1], bx2 = pb1[0], bx3 = pb1[1];
    v4f yv[4][4];
#pragma unroll
    for (int j = 0; j < 4; ++j) {
      if (FULLN || cbase + j * 16 < N) {           // scalar guard
        const v4f* pb = (const v4f*)(Wr[j] + k0 + 8 * half);
#pragma unroll
        for (int q = 0; q < 4; ++q) yv[j][q] = pb[(q & 1) + 4 * (q >> 1)];
      }
    }
    // ---- convert + 8 WMMAs ----
    v16h a0 = cvt16(ax0, ax1, ax2, ax3);
    v16h a1 = cvt16(bx0, bx1, bx2, bx3);
#pragma unroll
    for (int j = 0; j < 4; ++j) {
      if (FULLN || cbase + j * 16 < N) {
        v16h b = cvt16(yv[j][0], yv[j][1], yv[j][2], yv[j][3]);
        accA[j] = __builtin_amdgcn_wmma_f32_16x16x32_f16(false, a0, false, b,
                                                         (short)0, accA[j], false, false);
        accB[j] = __builtin_amdgcn_wmma_f32_16x16x32_f16(false, a1, false, b,
                                                         (short)0, accB[j], false, false);
      }
    }
  }
  if (!FULLN && k0 < K) {   // K tail: address-clamped loads + select, no branches
    v16h a0, a1;
#pragma unroll
    for (int e = 0; e < 16; ++e) {
      int k  = k0 + (e < 8 ? e : e + 8) + 8 * half;
      int kc = (k < K) ? k : (K - 1);
      float v0 = Arow0[kc], v1 = Arow1[kc];
      a0[e] = (k < K) ? (_Float16)v0 : (_Float16)0.0f;
      a1[e] = (k < K) ? (_Float16)v1 : (_Float16)0.0f;
    }
#pragma unroll
    for (int j = 0; j < 4; ++j) {
      if (cbase + j * 16 < N) {
        v16h b;
#pragma unroll
        for (int e = 0; e < 16; ++e) {
          int k  = k0 + (e < 8 ? e : e + 8) + 8 * half;
          int kc = (k < K) ? k : (K - 1);
          float v = Wr[j][kc];
          b[e] = (k < K) ? (_Float16)v : (_Float16)0.0f;
        }
        accA[j] = __builtin_amdgcn_wmma_f32_16x16x32_f16(false, a0, false, b,
                                                         (short)0, accA[j], false, false);
        accB[j] = __builtin_amdgcn_wmma_f32_16x16x32_f16(false, a1, false, b,
                                                         (short)0, accB[j], false, false);
      }
    }
  }
  // ---- epilogue: bias / residual / GELU / store (row-guarded) ----
#pragma unroll
  for (int j = 0; j < 4; ++j) {
    if (FULLN || cbase + j * 16 < N) {
      int n = cbase + j * 16 + l16;
      float bn = bias ? bias[n] : 0.0f;
#pragma unroll
      for (int mi = 0; mi < 2; ++mi) {
        v8f acc = mi ? accB[j] : accA[j];
#pragma unroll
        for (int r = 0; r < 8; ++r) {
          int m = tm * 32 + mi * 16 + r + 8 * half;
          if (m < M) {
            float v = acc[r] + bn;
            if (res) v += res[(size_t)m * ldR + n];
            if (act) v = gelu_exact(v);
            Out[(size_t)m * ldO + n] = v;
          }
        }
      }
    }
  }
}

// -------------------------- LayerNorm (wave per row) -----------------------
__global__ __launch_bounds__(256) void k_layernorm(
    const float* __restrict__ in, float* __restrict__ out,
    const float* __restrict__ g, const float* __restrict__ b, int T, int C)
{
  int row  = blockIdx.x * 8 + __builtin_amdgcn_readfirstlane(threadIdx.x >> 5);
  int lane = threadIdx.x & 31;
  if (row >= T) return;                           // scalar branch
  const float* x = in + (size_t)row * C;
  float lv[24];                                   // C <= 768
  int n = 0; float s = 0.0f;
  for (int c = lane; c < C; c += 32) { float v = x[c]; lv[n++] = v; s += v; }
  s = wave_sum(s);
  float m = s / (float)C;
  float vs = 0.0f;
  for (int i = 0; i < n; ++i) { float d = lv[i] - m; vs += d * d; }
  vs = wave_sum(vs);
  float r = rsqrtf(vs / (float)C + 1e-5f);
  float* o = out + (size_t)row * C;
  n = 0;
  for (int c = lane; c < C; c += 32) { o[c] = (lv[n++] - m) * r * g[c] + b[c]; }
}

// -------------------- window partition / unpartition -----------------------
// xw layout: [B*NW, 49, C]; roll(+shift) fused on partition, roll(-ushift)
// fused on unpartition (faithful asymmetric unshift: shift=3, unshift=4).
__global__ void k_window_part(const float* __restrict__ h, float* __restrict__ xw,
                              int B, int H, int W, int C, int shift)
{
  size_t total = (size_t)B * H * W * C;
  size_t i = blockIdx.x * (size_t)blockDim.x + threadIdx.x;
  if (i >= total) return;
  int c = (int)(i % C); size_t t = i / C;
  int tt = (int)(t % 49); size_t wn = t / 49;
  int nww = W / 7, nwh = H / 7;
  int wx = (int)(wn % nww); size_t r2 = wn / nww;
  int wy = (int)(r2 % nwh); int b = (int)(r2 / nwh);
  int y = wy * 7 + tt / 7, x = wx * 7 + tt % 7;
  int ys = (y - shift + H) % H, xs = (x - shift + W) % W;
  xw[i] = h[(((size_t)b * H + ys) * W + xs) * C + c];
}

__global__ void k_window_unpart(const float* __restrict__ xw, float* __restrict__ h,
                                int B, int H, int W, int C, int ushift)
{
  size_t total = (size_t)B * H * W * C;
  size_t i = blockIdx.x * (size_t)blockDim.x + threadIdx.x;
  if (i >= total) return;
  int c = (int)(i % C); size_t t = i / C;
  int tt = (int)(t % 49); size_t wn = t / 49;
  int nww = W / 7, nwh = H / 7;
  int wx = (int)(wn % nww); size_t r2 = wn / nww;
  int wy = (int)(r2 % nwh); int b = (int)(r2 / nwh);
  int y = wy * 7 + tt / 7, x = wx * 7 + tt % 7;
  int yd = (y - ushift + H) % H, xd = (x - ushift + W) % W;
  h[(((size_t)b * H + yd) * W + xd) * C + c] = xw[i];
}

// ----- attention core (per window, per head): scores + pos bias, no softmax
// qkv row layout: o = 3*c + {0:q,1:k,2:v}
__global__ __launch_bounds__(256) void k_attn(
    const float* __restrict__ qkv, float* __restrict__ cat,
    int C, int nh, int head, float inv_scale)
{
  __shared__ float att[49 * 49];
  int w = blockIdx.x;
  const float* base = qkv + (size_t)w * 49 * 3 * C;
  for (int idx = threadIdx.x; idx < 49 * 49; idx += blockDim.x) {
    int t = idx / 49, s = idx % 49;
    const float* q = base + (size_t)t * 3 * C;
    const float* k = base + (size_t)s * 3 * C;
    float d = 0.0f;
    for (int c = 0; c < C; ++c) d += q[3 * c] * k[3 * c + 1];
    int ty = t / 7, tx = t % 7, sy = s / 7, sx = s % 7;
    att[idx] = d * inv_scale + (float)((sx - tx + 6) + 13 * (sy - ty + 6));
  }
  __syncthreads();
  int ldc = nh * C;
  for (int idx = threadIdx.x; idx < 49 * C; idx += blockDim.x) {
    int t = idx / C, c = idx % C;
    const float* a = att + t * 49;
    float o = 0.0f;
    for (int s = 0; s < 49; ++s) o += a[s] * base[(size_t)s * 3 * C + 3 * c + 2];
    cat[((size_t)(w * 49 + t)) * ldc + (size_t)head * C + c] = o;
  }
}

// -------- patch rearrange: 'b c (h 4)(w 4) -> b h w (p0 p1 c)' --------------
__global__ void k_patch(const float* __restrict__ xin, float* __restrict__ xp)
{
  size_t total = (size_t)4 * 56 * 56 * 48;
  size_t i = blockIdx.x * (size_t)blockDim.x + threadIdx.x;
  if (i >= total) return;
  int col = (int)(i % 48); size_t t = i / 48;
  int x = (int)(t % 56); size_t r2 = t / 56;
  int y = (int)(r2 % 56); int b = (int)(r2 / 56);
  int p0 = col / 12, rr = col % 12, p1 = rr / 3, c = rr % 3;
  xp[i] = xin[(((size_t)b * 3 + c) * 224 + (4 * y + p0)) * 224 + (4 * x + p1)];
}

// -------- merge: 'b (h 2)(w 2) c -> b h w (c 2 2)' (channel-major) ---------
__global__ void k_merge(const float* __restrict__ h, float* __restrict__ xm,
                        int B, int H, int W, int C)
{
  int H2 = H / 2, W2 = W / 2, C4 = 4 * C;
  size_t total = (size_t)B * H2 * W2 * C4;
  size_t i = blockIdx.x * (size_t)blockDim.x + threadIdx.x;
  if (i >= total) return;
  int col = (int)(i % C4); size_t t = i / C4;
  int x = (int)(t % W2); size_t r2 = t / W2;
  int y = (int)(r2 % H2); int b = (int)(r2 / H2);
  int c = col >> 2, di = (col >> 1) & 1, dj = col & 1;
  xm[i] = h[(((size_t)b * H + 2 * y + di) * W + 2 * x + dj) * C + c];
}

// --------- token [B,H,W,C] -> NCHW, written at channel offset --------------
__global__ void k_tok2nchw(const float* __restrict__ tok, float* __restrict__ out,
                           int B, int H, int W, int C, int coff, int Ctot)
{
  size_t total = (size_t)B * C * H * W;
  size_t i = blockIdx.x * (size_t)blockDim.x + threadIdx.x;
  if (i >= total) return;
  int x = (int)(i % W); size_t r = i / W;
  int y = (int)(r % H); r /= H;
  int c = (int)(r % C); int b = (int)(r / C);
  out[(((size_t)b * Ctot + coff + c) * H + y) * W + x] =
      tok[(((size_t)b * H + y) * W + x) * C + c];
}

// --------- NCHW sub-tensor copy at channel offset (for x0 concat) ----------
__global__ void k_nchw_inject(const float* __restrict__ in, float* __restrict__ out,
                              int B, int C, int HW, int coff, int Ctot)
{
  size_t total = (size_t)B * C * HW;
  size_t i = blockIdx.x * (size_t)blockDim.x + threadIdx.x;
  if (i >= total) return;
  int p = (int)(i % HW); size_t r = i / HW;
  int c = (int)(r % C); int b = (int)(r / C);
  out[((size_t)b * Ctot + coff + c) * HW + p] = in[i];
}

// --------- ConvTranspose2d k=4 s=2 p=1, w[I,O,4,4], out at ch offset -------
__global__ void k_convT(const float* __restrict__ in, const float* __restrict__ w,
                        float* __restrict__ out, int B, int I, int O,
                        int Hin, int Win, int Ctot, int coff)
{
  int Ho = 2 * Hin, Wo = 2 * Win;
  size_t total = (size_t)B * O * Ho * Wo;
  for (size_t i = blockIdx.x * (size_t)blockDim.x + threadIdx.x; i < total;
       i += (size_t)gridDim.x * blockDim.x) {
    int ox = (int)(i % Wo); size_t r = i / Wo;
    int oy = (int)(r % Ho); r /= Ho;
    int o = (int)(r % O); int b = (int)(r / O);
    int kys[2], iys[2], nky = 0;
    for (int ky = (oy + 1) & 1; ky < 4; ky += 2) {
      int iy = (oy + 1 - ky) >> 1;
      if (iy >= 0 && iy < Hin) { kys[nky] = ky; iys[nky] = iy; ++nky; }
    }
    int kxs[2], ixs[2], nkx = 0;
    for (int kx = (ox + 1) & 1; kx < 4; kx += 2) {
      int ix = (ox + 1 - kx) >> 1;
      if (ix >= 0 && ix < Win) { kxs[nkx] = kx; ixs[nkx] = ix; ++nkx; }
    }
    float s = 0.0f;
    for (int ic = 0; ic < I; ++ic) {
      const float* ip = in + ((size_t)b * I + ic) * Hin * Win;
      const float* wp = w + ((size_t)ic * O + o) * 16;
      for (int a = 0; a < nky; ++a)
        for (int bb = 0; bb < nkx; ++bb)
          s += ip[iys[a] * Win + ixs[bb]] * wp[kys[a] * 4 + kxs[bb]];
    }
    out[(((size_t)b * Ctot + coff + o) * Ho + oy) * Wo + ox] = s;
  }
}

// ------------------- GroupNorm (groups=4): reduce + apply ------------------
__global__ __launch_bounds__(256) void k_gn_reduce(
    const float* __restrict__ x, float* __restrict__ stat,
    int C, int HW, int Ctot, int coff)
{
  int bg = blockIdx.x;                 // b*4 + g
  int g = bg & 3, b = bg >> 2;
  int cpg = C >> 2;
  size_t n = (size_t)cpg * HW;
  const float* base = x + ((size_t)b * Ctot + coff + (size_t)g * cpg) * HW;
  float s = 0.0f, s2 = 0.0f;
  for (size_t i = threadIdx.x; i < n; i += 256) {
    float v = base[i];                 // group channels are contiguous
    s += v; s2 += v * v;
  }
  __shared__ float sh[512];
  int tid = threadIdx.x;
  sh[tid] = s; sh[256 + tid] = s2;
  __syncthreads();
  for (int o = 128; o > 0; o >>= 1) {
    if (tid < o) { sh[tid] += sh[tid + o]; sh[256 + tid] += sh[256 + tid + o]; }
    __syncthreads();
  }
  if (tid == 0) {
    float m = sh[0] / (float)n;
    float var = sh[256] / (float)n - m * m;
    stat[bg * 2] = m;
    stat[bg * 2 + 1] = rsqrtf(var + 1e-5f);
  }
}

__global__ void k_gn_apply(float* __restrict__ x, const float* __restrict__ stat,
                           const float* __restrict__ g, const float* __restrict__ bta,
                           int B, int C, int HW, int Ctot, int coff, int act)
{
  size_t total = (size_t)B * C * HW;
  for (size_t i = blockIdx.x * (size_t)blockDim.x + threadIdx.x; i < total;
       i += (size_t)gridDim.x * blockDim.x) {
    int p = (int)(i % HW); size_t r = i / HW;
    int c = (int)(r % C); int b = (int)(r / C);
    int grp = c / (C >> 2);
    float m = stat[(b * 4 + grp) * 2], rs = stat[(b * 4 + grp) * 2 + 1];
    size_t o = ((size_t)b * Ctot + coff + c) * HW + p;
    float v = (x[o] - m) * rs * g[c] + bta[c];
    if (act) v = gelu_exact(v);
    x[o] = v;
  }
}

// ------------------- Conv2d 3x3 s=1 p=1 (no bias), w[O,I,3,3] --------------
__global__ void k_conv3(const float* __restrict__ in, const float* __restrict__ w,
                        float* __restrict__ out, int B, int I, int O, int H, int W)
{
  size_t total = (size_t)B * O * H * W;
  for (size_t i = blockIdx.x * (size_t)blockDim.x + threadIdx.x; i < total;
       i += (size_t)gridDim.x * blockDim.x) {
    int x = (int)(i % W); size_t r = i / W;
    int y = (int)(r % H); r /= H;
    int o = (int)(r % O); int b = (int)(r / O);
    float s = 0.0f;
    for (int ic = 0; ic < I; ++ic) {
      const float* ip = in + ((size_t)b * I + ic) * H * W;
      const float* wp = w + ((size_t)o * I + ic) * 9;
      for (int dy = -1; dy <= 1; ++dy) {
        int yy = y + dy; if (yy < 0 || yy >= H) continue;
        for (int dx = -1; dx <= 1; ++dx) {
          int xx = x + dx; if (xx < 0 || xx >= W) continue;
          s += ip[yy * W + xx] * wp[(dy + 1) * 3 + (dx + 1)];
        }
      }
    }
    out[i] = s;
  }
}

// ------------------- softmax over 3 channels (axis=1) ----------------------
__global__ void k_softmax3(float* __restrict__ x, int HW)
{
  size_t total = (size_t)4 * HW;
  size_t i = blockIdx.x * (size_t)blockDim.x + threadIdx.x;
  if (i >= total) return;
  int p = (int)(i % HW); int b = (int)(i / HW);
  float* px = x + (size_t)b * 3 * HW + p;
  float a = px[0], c = px[HW], d = px[2 * HW];
  float m = fmaxf(a, fmaxf(c, d));
  float ea = expf(a - m), eb = expf(c - m), ec = expf(d - m);
  float inv = 1.0f / (ea + eb + ec);
  px[0] = ea * inv; px[HW] = eb * inv; px[2 * HW] = ec * inv;
}

// ===========================================================================
// Host side
// ===========================================================================
struct Blk {
  const float *ln_g, *ln_b, *qkv_w, *qkv_b, *mhp_w, *mhp_b;
  const float *mlp_ln_g, *mlp_ln_b, *fc1_w, *fc1_b, *fc2_w, *fc2_b;
};
struct UpP { const float *w, *g, *b; };
struct Params {
  const float *x, *embed_w, *embed_b;
  Blk s1[2], s2[2], s3[6], s4[2];
  const float *m2w, *m2b, *m3w, *m3b, *m4w, *m4b;
  UpP u0, u1, u2;
  const float *u3w1, *u3g1, *u3b1, *u3w2, *u3g2, *u3b2;
  const float *ow1, *og1, *ob1, *ow2;
};

static void fill_insertion(void* const* d, Params& P) {
  int i = 0;
  auto f = [&]() { return (const float*)d[i++]; };
  P.x = f(); P.embed_w = f(); P.embed_b = f();
  auto blk = [&](Blk& b) {
    b.ln_g = f(); b.ln_b = f(); b.qkv_w = f(); b.qkv_b = f();
    b.mhp_w = f(); b.mhp_b = f(); b.mlp_ln_g = f(); b.mlp_ln_b = f();
    b.fc1_w = f(); b.fc1_b = f(); b.fc2_w = f(); b.fc2_b = f();
  };
  blk(P.s1[0]); blk(P.s1[1]);
  P.m2w = f(); P.m2b = f();
  blk(P.s2[0]); blk(P.s2[1]);
  P.m3w = f(); P.m3b = f();
  for (int k = 0; k < 6; ++k) blk(P.s3[k]);
  P.m4w = f(); P.m4b = f();
  blk(P.s4[0]); blk(P.s4[1]);
  P.u0.w = f(); P.u0.g = f(); P.u0.b = f();
  P.u1.w = f(); P.u1.g = f(); P.u1.b = f();
  P.u2.w = f(); P.u2.g = f(); P.u2.b = f();
  P.u3w1 = f(); P.u3g1 = f(); P.u3b1 = f();
  P.u3w2 = f(); P.u3g2 = f(); P.u3b2 = f();
  P.ow1 = f(); P.og1 = f(); P.ob1 = f(); P.ow2 = f();
}

// jax tree_flatten sorts dict keys alphabetically ('params' < 'x', etc.)
static void fill_sorted(void* const* d, Params& P) {
  int i = 0;
  auto f = [&]() { return (const float*)d[i++]; };
  P.embed_b = f(); P.embed_w = f();
  P.m2b = f(); P.m2w = f(); P.m3b = f(); P.m3w = f(); P.m4b = f(); P.m4w = f();
  P.ob1 = f(); P.og1 = f(); P.ow1 = f(); P.ow2 = f();
  auto blk = [&](Blk& b) {
    b.fc1_b = f(); b.fc1_w = f(); b.fc2_b = f(); b.fc2_w = f();
    b.ln_b = f(); b.ln_g = f(); b.mhp_b = f(); b.mhp_w = f();
    b.mlp_ln_b = f(); b.mlp_ln_g = f(); b.qkv_b = f(); b.qkv_w = f();
  };
  blk(P.s1[0]); blk(P.s1[1]);
  blk(P.s2[0]); blk(P.s2[1]);
  for (int k = 0; k < 6; ++k) blk(P.s3[k]);
  blk(P.s4[0]); blk(P.s4[1]);
  P.u0.b = f(); P.u0.g = f(); P.u0.w = f();
  P.u1.b = f(); P.u1.g = f(); P.u1.w = f();
  P.u2.b = f(); P.u2.g = f(); P.u2.w = f();
  P.u3b1 = f(); P.u3b2 = f(); P.u3g1 = f(); P.u3g2 = f();
  P.u3w1 = f(); P.u3w2 = f();
  P.x = f();
}

static inline int gs(size_t n) { return (int)((n + 255) / 256); }

static inline void gemm(hipStream_t st, const float* A, const float* Wt,
                        const float* bias, const float* res, float* Out,
                        int M, int N, int K, int ldA, int ldO, int ldR, int act) {
  int tiles = ((M + 31) / 32) * ((N + 63) / 64);
  int blocks = (tiles + 7) / 8;
  if ((N & 63) == 0)
    k_gemm_wmma<true><<<blocks, 256, 0, st>>>(A, Wt, bias, res, Out,
                                              M, N, K, ldA, ldO, ldR, act);
  else
    k_gemm_wmma<false><<<blocks, 256, 0, st>>>(A, Wt, bias, res, Out,
                                               M, N, K, ldA, ldO, ldR, act);
}

static void run_block(hipStream_t st, const Blk& p, float* h, float* xw, float* xn,
                      float* qkvb, float* cat, float* t1, float* mlp,
                      int H, int C, bool shift) {
  const int B = 4, nh = 4;
  int W = H;
  int NW = B * (H / 7) * (W / 7);
  int T = NW * 49;
  float inv_scale = (float)(1.0 / sqrt((double)C));
  size_t tot = (size_t)B * H * W * C;
  k_window_part<<<gs(tot), 256, 0, st>>>(h, xw, B, H, W, C, shift ? 3 : 0);
  for (int hd = 0; hd < nh; ++hd) {
    k_layernorm<<<(T + 7) / 8, 256, 0, st>>>(xw, xn, p.ln_g + (size_t)hd * C,
                                             p.ln_b + (size_t)hd * C, T, C);
    gemm(st, xn, p.qkv_w + (size_t)hd * 3 * C * C, p.qkv_b + (size_t)hd * 3 * C,
         nullptr, qkvb, T, 3 * C, C, C, 3 * C, 0, 0);
    k_attn<<<NW, 256, 0, st>>>(qkvb, cat, C, nh, hd, inv_scale);
  }
  gemm(st, cat, p.mhp_w, p.mhp_b, xw, t1, T, C, nh * C, nh * C, C, C, 0);
  k_layernorm<<<(T + 7) / 8, 256, 0, st>>>(t1, xn, p.mlp_ln_g, p.mlp_ln_b, T, C);
  gemm(st, xn, p.fc1_w, p.fc1_b, nullptr, mlp, T, 4 * C, C, C, 4 * C, 0, 1);
  gemm(st, mlp, p.fc2_w, p.fc2_b, t1, xw, T, C, 4 * C, 4 * C, C, C, 0);
  k_window_unpart<<<gs(tot), 256, 0, st>>>(xw, h, B, H, W, C, shift ? 4 : 0);
}

extern "C" void kernel_launch(void* const* d_in, const int* in_sizes, int n_in,
                              void* d_out, int out_size, void* d_ws, size_t ws_size,
                              hipStream_t stream) {
  (void)n_in; (void)out_size; (void)ws_size;
  Params P;
  if (in_sizes[0] == 4 * 3 * 224 * 224) fill_insertion(d_in, P);
  else                                  fill_sorted(d_in, P);

  float* ws = (float*)d_ws;
  // transformer arenas (float offsets); ~237 MB total workspace
  float* h    = ws + 0;          // 1,204,224
  float* xw   = ws + 1204224;    // 1,204,224
  float* xn   = ws + 2408448;    // 1,204,224
  float* qkv  = ws + 3612672;    // 3,612,672
  float* cat  = ws + 7225344;    // 4,816,896  (also xp / merge scratch)
  float* t1   = ws + 12042240;   // 1,204,224
  float* mlp  = ws + 13246464;   // 4,816,896  -> region end 18,063,360
  float* x1t  = ws + 18063360;   // 1,204,224
  float* x2t  = x1t + 1204224;   //   602,112
  float* x3t  = x2t + 602112;    //   301,056
  float* gns  = x3t + 301056;    //        64
  float* cat3 = gns + 64;        // 19,869,696
  float* c1   = cat3 + 19869696; // 19,267,584
  // decoder overlays (transformer arenas are dead by then)
  float* cat0 = ws + 0;          //   602,112  [B,768,14,14]
  float* cat1 = cat0 + 602112;   // 1,806,336  [B,576,28,28]
  float* cat2 = cat1 + 1806336;  // 4,816,896  [B,384,56,56]
  float* u3a  = cat2 + 4816896;  // 9,633,792  [B,192,112,112]
  float* d0   = u3a + 9633792;   //   150,528  [B,768,7,7]
  hipStream_t st = stream;

  // ---- patch embed -------------------------------------------------------
  k_patch<<<gs((size_t)602112), 256, 0, st>>>(P.x, cat);
  gemm(st, cat, P.embed_w, P.embed_b, nullptr, h, 12544, 96, 48, 48, 96, 0, 0);

  // ---- stage 1 (C=96, 56x56) --------------------------------------------
  for (int i = 0; i < 2; ++i)
    run_block(st, P.s1[i], h, xw, xn, qkv, cat, t1, mlp, 56, 96, (i % 2) == 1);
  hipMemcpyAsync(x1t, h, sizeof(float) * 1204224, hipMemcpyDeviceToDevice, st);
  k_merge<<<gs((size_t)3136 * 384), 256, 0, st>>>(h, cat, 4, 56, 56, 96);
  gemm(st, cat, P.m2w, P.m2b, nullptr, h, 3136, 192, 384, 384, 192, 0, 0);

  // ---- stage 2 (C=192, 28x28) -------------------------------------------
  for (int i = 0; i < 2; ++i)
    run_block(st, P.s2[i], h, xw, xn, qkv, cat, t1, mlp, 28, 192, (i % 2) == 1);
  hipMemcpyAsync(x2t, h, sizeof(float) * 602112, hipMemcpyDeviceToDevice, st);
  k_merge<<<gs((size_t)784 * 768), 256, 0, st>>>(h, cat, 4, 28, 28, 192);
  gemm(st, cat, P.m3w, P.m3b, nullptr, h, 784, 384, 768, 768, 384, 0, 0);

  // ---- stage 3 (C=384, 14x14) -------------------------------------------
  for (int i = 0; i < 6; ++i)
    run_block(st, P.s3[i], h, xw, xn, qkv, cat, t1, mlp, 14, 384, (i % 2) == 1);
  hipMemcpyAsync(x3t, h, sizeof(float) * 301056, hipMemcpyDeviceToDevice, st);
  k_merge<<<gs((size_t)196 * 1536), 256, 0, st>>>(h, cat, 4, 14, 14, 384);
  gemm(st, cat, P.m4w, P.m4b, nullptr, h, 196, 768, 1536, 1536, 768, 0, 0);

  // ---- stage 4 (C=768, 7x7) ---------------------------------------------
  for (int i = 0; i < 2; ++i)
    run_block(st, P.s4[i], h, xw, xn, qkv, cat, t1, mlp, 7, 768, (i % 2) == 1);

  // ---- decoder -----------------------------------------------------------
  k_tok2nchw<<<gs((size_t)4 * 768 * 49), 256, 0, st>>>(h, d0, 4, 7, 7, 768, 0, 768);
  // up0: 768 -> 384 @14, concat x3 (384) -> cat0[768]
  k_convT<<<gs((size_t)4 * 384 * 196), 256, 0, st>>>(d0, P.u0.w, cat0, 4, 768, 384, 7, 7, 768, 0);
  k_gn_reduce<<<16, 256, 0, st>>>(cat0, gns, 384, 196, 768, 0);
  k_gn_apply<<<gs((size_t)4 * 384 * 196), 256, 0, st>>>(cat0, gns, P.u0.g, P.u0.b, 4, 384, 196, 768, 0, 1);
  k_tok2nchw<<<gs((size_t)4 * 384 * 196), 256, 0, st>>>(x3t, cat0, 4, 14, 14, 384, 384, 768);
  // up1: 768 -> 384 @28, concat x2 (192) -> cat1[576]
  k_convT<<<gs((size_t)4 * 384 * 784), 256, 0, st>>>(cat0, P.u1.w, cat1, 4, 768, 384, 14, 14, 576, 0);
  k_gn_reduce<<<16, 256, 0, st>>>(cat1, gns, 384, 784, 576, 0);
  k_gn_apply<<<gs((size_t)4 * 384 * 784), 256, 0, st>>>(cat1, gns, P.u1.g, P.u1.b, 4, 384, 784, 576, 0, 1);
  k_tok2nchw<<<gs((size_t)4 * 192 * 784), 256, 0, st>>>(x2t, cat1, 4, 28, 28, 192, 384, 576);
  // up2: 576 -> 288 @56, concat x1 (96) -> cat2[384]
  k_convT<<<gs((size_t)4 * 288 * 3136), 256, 0, st>>>(cat1, P.u2.w, cat2, 4, 576, 288, 28, 28, 384, 0);
  k_gn_reduce<<<16, 256, 0, st>>>(cat2, gns, 288, 3136, 384, 0);
  k_gn_apply<<<gs((size_t)4 * 288 * 3136), 256, 0, st>>>(cat2, gns, P.u2.g, P.u2.b, 4, 288, 3136, 384, 0, 1);
  k_tok2nchw<<<gs((size_t)4 * 96 * 3136), 256, 0, st>>>(x1t, cat2, 4, 56, 56, 96, 288, 384);
  // up3a: 384 -> 192 @112
  k_convT<<<gs((size_t)4 * 192 * 12544), 256, 0, st>>>(cat2, P.u3w1, u3a, 4, 384, 192, 56, 56, 192, 0);
  k_gn_reduce<<<16, 256, 0, st>>>(u3a, gns, 192, 12544, 192, 0);
  k_gn_apply<<<gs((size_t)4 * 192 * 12544), 256, 0, st>>>(u3a, gns, P.u3g1, P.u3b1, 4, 192, 12544, 192, 0, 1);
  // up3b: 192 -> 96 @224, concat x0 (3) -> cat3[99]
  k_convT<<<gs((size_t)4 * 96 * 50176), 256, 0, st>>>(u3a, P.u3w2, cat3, 4, 192, 96, 112, 112, 99, 0);
  k_gn_reduce<<<16, 256, 0, st>>>(cat3, gns, 96, 50176, 99, 0);
  k_gn_apply<<<gs((size_t)4 * 96 * 50176), 256, 0, st>>>(cat3, gns, P.u3g2, P.u3b2, 4, 96, 50176, 99, 0, 1);
  k_nchw_inject<<<gs((size_t)4 * 3 * 50176), 256, 0, st>>>(P.x, cat3, 4, 3, 50176, 96, 99);
  // out head: conv3(99->96) + GN + GELU, conv3(96->3), softmax(ch)
  k_conv3<<<gs((size_t)4 * 96 * 50176), 256, 0, st>>>(cat3, P.ow1, c1, 4, 99, 96, 224, 224);
  k_gn_reduce<<<16, 256, 0, st>>>(c1, gns, 96, 50176, 96, 0);
  k_gn_apply<<<gs((size_t)4 * 96 * 50176), 256, 0, st>>>(c1, gns, P.og1, P.ob1, 4, 96, 50176, 96, 0, 1);
  k_conv3<<<gs((size_t)4 * 3 * 50176), 256, 0, st>>>(c1, P.ow2, (float*)d_out, 4, 96, 3, 224, 224);
  k_softmax3<<<gs((size_t)4 * 50176), 256, 0, st>>>((float*)d_out, 50176);
}
